// VanillaRNN_50122268345139
// MI455X (gfx1250) — matrix operand order repaired and checked by reading, compiled, not run on hardware
//
#include <hip/hip_runtime.h>

// ---------------- problem constants ----------------
#define B_SZ 2048
#define T_SZ 1024
#define H    32
#define NA   4      // n_actions (output width, padded to 16 in WMMA)
#define IND  5      // input dim = 4 actions + 1 reward (padded to K=32)

#define WAVES_PER_BLOCK 4
#define ROWS_PER_WAVE   16
#define LDS_ROW_HALFS   40   // 80B row stride: 16B-aligned, bank-conflict padded

typedef __attribute__((ext_vector_type(16))) _Float16 v16h;
typedef __attribute__((ext_vector_type(8)))  _Float16 v8h;
typedef __attribute__((ext_vector_type(8)))  float    v8f;

// same-wave LDS store->load ordering fence (hardware keeps DS in order per
// wave; this stops the compiler reordering the cross-lane RAW)
__device__ __forceinline__ void wave_lds_fence() {
    asm volatile("s_wait_dscnt 0" ::: "memory");
}

// tanh on the recurrent critical path. CDNA5 lists TANH as a TRANS32 op;
// prefer the single-instruction form if the toolchain exposes it.
__device__ __forceinline__ float fast_tanh(float x) {
#if __has_builtin(__builtin_amdgcn_tanhf)
    return __builtin_amdgcn_tanhf(x);                      // v_tanh_f32
#elif __has_builtin(__builtin_amdgcn_tanh_f32)
    return __builtin_amdgcn_tanh_f32(x);                   // v_tanh_f32
#else
    float e = __expf(2.0f * x);                            // v_exp_f32
    return 1.0f - 2.0f * __builtin_amdgcn_rcpf(e + 1.0f);  // inf->1, 0->-1
#endif
}

// Load A-operand (16x32 f16, row-major in LDS) in contiguous k order:
// lane l: M = l&15, halves 0..15 <- K = base..base+15
// with base = (l>=16) ? 16 : 0.  Row-major memory => two plain 16B loads.
__device__ __forceinline__ v16h load_a_rowmajor(const _Float16* rowK0) {
    v8h lo = *(const v8h*)(rowK0);
    v8h hi = *(const v8h*)(rowK0 + 8);
    v16h a;
#pragma unroll
    for (int j = 0; j < 8; ++j) { a[j] = lo[j]; a[j + 8] = hi[j]; }
    return a;
}

__device__ __forceinline__ v8f wmma_f16(v16h a, v16h b, v8f c) {
    return __builtin_amdgcn_wmma_f32_16x16x32_f16(
        /*neg_a=*/false, a, /*neg_b=*/false, b,
        /*c_mod=*/(short)0, c, /*reuse_a=*/false, /*reuse_b=*/false);
}

__global__ __launch_bounds__(WAVES_PER_BLOCK * 32, 1)
void rnn_scan_wmma(const float* __restrict__ actions,   // [B,T,4]
                   const float* __restrict__ rewards,   // [B,T,1]
                   const float* __restrict__ W_ih,      // [32,5]
                   const float* __restrict__ W_hh,      // [32,32]
                   const float* __restrict__ b_ih,      // [32]
                   const float* __restrict__ b_hh,      // [32]
                   const float* __restrict__ W_ro,      // [4,32]
                   const float* __restrict__ b_ro,      // [4]
                   float* __restrict__ out_logits,      // [B,T,4]
                   float* __restrict__ out_hT)          // [B,32]
{
    __shared__ __align__(16) _Float16 hbuf[WAVES_PER_BLOCK][ROWS_PER_WAVE][LDS_ROW_HALFS];

    const int lane = threadIdx.x & 31;
    const int wave = threadIdx.x >> 5;
    const int nl   = lane & 15;        // column within 16-wide tile (N for B/C/D, M for A)
    const int hi   = lane >> 4;        // half-wave selector
    const int m0   = (blockIdx.x * WAVES_PER_BLOCK + wave) * ROWS_PER_WAVE;

    // ---------------- loop-invariant B operands (contiguous k order) ----------------
    // B layout (32x16, 16-bit): lane l holds column n = l&15; half j -> K = 16*hi + j.
    // W_hh^T tile:  B[k][n] = W_hh[n][k]  (contiguous row segment of W_hh)
    v16h bHH0, bHH1;
    {
        const int kb = hi * 16;
#pragma unroll
        for (int j = 0; j < 16; ++j) {
            bHH0[j] = (_Float16)W_hh[(nl      ) * H + kb + j];
            bHH1[j] = (_Float16)W_hh[(nl + 16 ) * H + kb + j];
        }
    }
    // W_ih^T tile, K padded 5->32: only hi==0 / j<5 nonzero
    v16h bIH0 = {}, bIH1 = {};
    if (hi == 0) {
#pragma unroll
        for (int j = 0; j < IND; ++j) {
            bIH0[j] = (_Float16)W_ih[(nl      ) * IND + j];
            bIH1[j] = (_Float16)W_ih[(nl + 16 ) * IND + j];
        }
    }
    // W_ro^T tile, N padded 4->16: column a = nl (valid a<4)
    v16h bRO = {};
    if (nl < NA) {
        const int kb = hi * 16;
#pragma unroll
        for (int j = 0; j < 16; ++j)
            bRO[j] = (_Float16)W_ro[nl * H + kb + j];
    }

    // Biases broadcast in C/D layout (element depends only on n = column = lane&15)
    const float bias0 = b_ih[nl]      + b_hh[nl];
    const float bias1 = b_ih[nl + 16] + b_hh[nl + 16];
    const float broV  = (nl < NA) ? b_ro[nl] : 0.0f;

    // A-layout LDS read pointer for this lane (row-major h tile)
    const _Float16* arow = &hbuf[wave][nl][hi * 16];

    // h_{-1} = 0 directly in A layout
    v16h aH = {};

    // per-lane x stream base (lane nl<16 owns batch row m0+nl)
    const size_t xrow = (size_t)(m0 + nl) * T_SZ;

    v8f c0 = {}, c1 = {};   // live-out: last hidden state (D layout, post-tanh)

    for (int t = 0; t < T_SZ; ++t) {
        // ---- x_t into A layout (K padded 5->32): only lanes 0..15, halves 0..4 ----
        v16h aX = {};
        if (hi == 0) {
            const float4 act = *(const float4*)(actions + (xrow + t) * 4);
            const float  rew = rewards[xrow + t];
            aX[0] = (_Float16)act.x;  aX[1] = (_Float16)act.y;
            aX[2] = (_Float16)act.z;  aX[3] = (_Float16)act.w;
            aX[4] = (_Float16)rew;
            if (t + 16 < T_SZ) {
                __builtin_prefetch(actions + (xrow + t + 16) * 4, 0, 3);
                __builtin_prefetch(rewards + (xrow + t + 16), 0, 3);
            }
        }

        // ---- pre-activation: bias + x@W_ih^T + h@W_hh^T  (4 WMMAs, K=32 each) ----
#pragma unroll
        for (int r = 0; r < 8; ++r) { c0[r] = bias0; c1[r] = bias1; }
        c0 = wmma_f16(aX, bIH0, c0);
        c1 = wmma_f16(aX, bIH1, c1);
        c0 = wmma_f16(aH, bHH0, c0);
        c1 = wmma_f16(aH, bHH1, c1);

        // ---- h_t = tanh(.) ----
#pragma unroll
        for (int r = 0; r < 8; ++r) { c0[r] = fast_tanh(c0[r]); c1[r] = fast_tanh(c1[r]); }

        // ---- D layout -> row-major f16 LDS tile (per-wave private) ----
#pragma unroll
        for (int r = 0; r < 8; ++r) {
            const int m = r + hi * 8;                 // D: VGPR r holds M = r (+8 for upper lanes)
            hbuf[wave][m][nl]      = (_Float16)c0[r];
            hbuf[wave][m][nl + 16] = (_Float16)c1[r];
        }
        wave_lds_fence();

        // ---- reload h_t in A layout (used by logits now AND recurrence next step) ----
        aH = load_a_rowmajor(arow);

        // ---- logits_t = h_t @ W_ro^T + b_ro  (1 WMMA, N padded to 16) ----
        v8f cl;
#pragma unroll
        for (int r = 0; r < 8; ++r) cl[r] = broV;
        cl = wmma_f16(aH, bRO, cl);

        if (nl < NA) {
#pragma unroll
            for (int r = 0; r < 8; ++r) {
                const int m = r + hi * 8;
                out_logits[((size_t)(m0 + m) * T_SZ + t) * NA + nl] = cl[r];
            }
        }
    }

    // ---- h_T [B,32] from last post-tanh state (D layout) ----
#pragma unroll
    for (int r = 0; r < 8; ++r) {
        const int m = r + hi * 8;
        out_hT[(size_t)(m0 + m) * H + nl]      = c0[r];
        out_hT[(size_t)(m0 + m) * H + nl + 16] = c1[r];
    }
}

extern "C" void kernel_launch(void* const* d_in, const int* in_sizes, int n_in,
                              void* d_out, int out_size, void* d_ws, size_t ws_size,
                              hipStream_t stream) {
    (void)in_sizes; (void)n_in; (void)out_size; (void)d_ws; (void)ws_size;
    const float* actions = (const float*)d_in[0];
    const float* rewards = (const float*)d_in[1];
    const float* W_ih    = (const float*)d_in[2];
    const float* W_hh    = (const float*)d_in[3];
    const float* b_ih    = (const float*)d_in[4];
    const float* b_hh    = (const float*)d_in[5];
    const float* W_ro    = (const float*)d_in[6];
    const float* b_ro    = (const float*)d_in[7];

    float* out_logits = (float*)d_out;                              // [B,T,4]
    float* out_hT     = out_logits + (size_t)B_SZ * T_SZ * NA;      // [B,32]

    dim3 grid(B_SZ / (WAVES_PER_BLOCK * ROWS_PER_WAVE));            // 32 blocks
    dim3 block(WAVES_PER_BLOCK * 32);                               // 4 waves (wave32)
    hipLaunchKernelGGL(rnn_scan_wmma, grid, block, 0, stream,
                       actions, rewards, W_ih, W_hh, b_ih, b_hh, W_ro, b_ro,
                       out_logits, out_hT);
}
